// GraphAttention_83846351552724
// MI455X (gfx1250) — matrix operand order repaired and checked by reading, not compiled, not run on hardware
//
#include <hip/hip_runtime.h>
#include <hip/hip_bf16.h>

#define B_DIM   4
#define N_PEP_  4096
#define N_REC_  10
#define EMB_    256
#define OUT_    256

// packed-weight layout (bf16 elements) inside d_ws:
//   Wp: [8 ks][16 tile][32 lane][16]  -> 65536 elems
//   Wr: [16 ks][16 tile][32 lane][16] -> 131072 elems
//   Wv: same as Wr                    -> 131072 elems
#define WP_OFF 0
#define WR_OFF 65536
#define WV_OFF 196608

// LDS A-tile row strides (bf16 elems), padded +8 so consecutive rows are 4
// banks apart -> conflict-free 16-lane ds_load_b128 fragment fetches.
#define RE_STRIDE 520   // 512 + 8
#define PQ_STRIDE 264   // 256 + 8

typedef __attribute__((ext_vector_type(16))) __bf16 v16bf;
typedef __attribute__((ext_vector_type(8)))  float  v8f;

__device__ __forceinline__ v8f wmma_bf16(v16bf a, v16bf b, v8f c) {
  // (neg_a, A, neg_b, B, c_mod, C, reuse_a, reuse_b)
  return __builtin_amdgcn_wmma_f32_16x16x32_bf16(false, a, false, b, (short)0, c,
                                                 false, false);
}

// A fragment from a bf16 LDS row: elems 0..15 = K[kb+half*16 .. +15]
// (contiguous k order, matching the packed B fragments)
__device__ __forceinline__ v16bf a_frag(const __bf16* rowkb, int half) {
  union { uint4 u[2]; v16bf v; } f;
  f.u[0] = *(const uint4*)(rowkb + half * 16);
  f.u[1] = *(const uint4*)(rowkb + half * 16 + 8);
  return f.v;
}

// B fragment: 32 contiguous bytes per lane from the packed-weight buffer.
__device__ __forceinline__ v16bf b_frag(const __bf16* p) {
  union { uint4 u[2]; v16bf v; } f;
  f.u[0] = ((const uint4*)p)[0];
  f.u[1] = ((const uint4*)p)[1];
  return f.v;
}

// ---------------- prelude: pack f32 weights -> bf16 B-fragment order --------
__global__ __launch_bounds__(256) void pack_weights_kernel(
    const float* __restrict__ Wp_w, const float* __restrict__ Wr_w,
    const float* __restrict__ Wv_w, __bf16* __restrict__ ws)
{
  int job = blockIdx.x * 256 + threadIdx.x;   // one job = 16 bf16 outputs
  if (job >= 20480) return;
  const float* src; __bf16* dst; int K; int j;
  if (job < 4096)       { src = Wp_w; dst = ws + WP_OFF; K = 256; j = job; }
  else if (job < 12288) { src = Wr_w; dst = ws + WR_OFF; K = 512; j = job - 4096; }
  else                  { src = Wv_w; dst = ws + WV_OFF; K = 512; j = job - 12288; }
  const int lane  = j & 31;
  const int tile  = (j >> 5) & 15;
  const int ks    = j >> 9;
  const int laneM = lane & 15, hh = lane >> 4;
  const float* p = src + (size_t)(tile * 16 + laneM) * K + ks * 32 + hh * 16;
  __bf16* q = dst + (size_t)j * 16;
#pragma unroll
  for (int i = 0; i < 16; ++i) q[i] = (__bf16)p[i];
}

// ---------------- main fused graph-attention kernel --------------------------
// 256 threads = 8 wave32; cap at <=256 VGPRs (4 waves/SIMD) to avoid spills
// and the s_set_vgpr_msb addressing overhead seen with full unroll.
__global__ __launch_bounds__(256, 4) void ga_wmma_kernel(
    const float* __restrict__ rec,  const float* __restrict__ pep,
    const float* __restrict__ edge, const __bf16* __restrict__ wpk,
    const float* __restrict__ Wp_b, const float* __restrict__ Wr_b,
    const float* __restrict__ Wv_b, float* __restrict__ out)
{
  __shared__ __align__(16) __bf16 lds_re[16 * RE_STRIDE];  // rec|edge tile (bf16)
  __shared__ __align__(16) __bf16 lds_pq[16 * PQ_STRIDE];  // pep tile (bf16)
  __shared__ float lds_logit[N_REC_][16];

  const int tid   = threadIdx.x;
  const int wave  = tid >> 5;
  const int lane  = tid & 31;
  const int laneM = lane & 15;     // A row / B-C-D col within tile
  const int half  = lane >> 4;

  const int bid = blockIdx.x;
  const int b   = bid >> 8;            // batch (N_PEP/16 == 256 tiles)
  const int n0  = (bid & 255) << 4;    // first pep row of this tile

  const int ot0 = wave * 2;            // two output-col tiles per wave
  const int ot1 = wave * 2 + 1;
  const int c0  = ot0 * 16, c1 = ot1 * 16;

  const __bf16* wr_pk = wpk + WR_OFF;
  const __bf16* wv_pk = wpk + WV_OFF;

  // stage pep tile (f32 -> bf16) + zero logit slots, one barrier
  const int srow = tid & 15, schunk = tid >> 4;
  {
    const float* p = pep + ((size_t)b * N_PEP_ + n0 + srow) * EMB_ + schunk * 16;
    __bf16* q = lds_pq + srow * PQ_STRIDE + schunk * 16;
#pragma unroll
    for (int i = 0; i < 16; ++i) q[i] = (__bf16)p[i];
  }
  if (tid < N_REC_ * 16) ((float*)lds_logit)[tid] = 0.0f;
  __syncthreads();

  // ---------------- pep_query = pep @ Wp^T + Wp_b (32 cols/wave) ------------
  v8f pq0 = {}; v8f pq1 = {};
#pragma unroll 2
  for (int ks = 0; ks < EMB_ / 32; ++ks) {
    v16bf a = a_frag(lds_pq + laneM * PQ_STRIDE + ks * 32, half);
    pq0 = wmma_bf16(a, b_frag(wpk + (size_t)(((ks * 16 + ot0) * 32 + lane) * 16)), pq0);
    pq1 = wmma_bf16(a, b_frag(wpk + (size_t)(((ks * 16 + ot1) * 32 + lane) * 16)), pq1);
  }
  {
    const float bb0 = Wp_b[c0 + laneM];
    const float bb1 = Wp_b[c1 + laneM];
#pragma unroll
    for (int i = 0; i < 8; ++i) { pq0[i] += bb0; pq1[i] += bb1; }
  }

  // ---------------- streaming (online-softmax) attention over neighbors -----
  float m[8], s[8];
  v8f o0 = {}; v8f o1 = {};
#pragma unroll
  for (int i = 0; i < 8; ++i) { m[i] = -3.0e38f; s[i] = 0.0f; }

  const float kb0 = Wr_b[c0 + laneM], kb1 = Wr_b[c1 + laneM];
  const float vb0 = Wv_b[c0 + laneM], vb1 = Wv_b[c1 + laneM];

  // staging source for this thread (row srow, 32-float chunk schunk)
  const size_t rowbase = ((size_t)b * N_PEP_ + n0 + srow) * (size_t)N_REC_ * EMB_;
  const float* sp_rec  = rec  + rowbase + (size_t)schunk * 32;
  const float* sp_edge = edge + rowbase + (size_t)(schunk - 8) * 32;

  for (int r = 0; r < N_REC_; ++r) {
    __syncthreads();  // all waves done reading lds_re from previous r
    {
      const float* p = (schunk < 8) ? (sp_rec + (size_t)r * EMB_)
                                    : (sp_edge + (size_t)r * EMB_);
      if (r + 1 < N_REC_) __builtin_prefetch(p + EMB_, 0, 0);  // global_prefetch_b8
      __bf16* q = lds_re + srow * RE_STRIDE + schunk * 32;
#pragma unroll
      for (int i = 0; i < 32; ++i) q[i] = (__bf16)p[i];
    }
    __syncthreads();  // tile visible to all waves

    v8f k0 = {}; v8f k1 = {}; v8f v0 = {}; v8f v1 = {};
#pragma unroll 2
    for (int ks = 0; ks < (2 * EMB_) / 32; ++ks) {   // K = 512 = concat(rec,edge)
      v16bf a = a_frag(lds_re + laneM * RE_STRIDE + ks * 32, half);
      const size_t w0 = (size_t)(((ks * 16 + ot0) * 32 + lane) * 16);
      const size_t w1 = (size_t)(((ks * 16 + ot1) * 32 + lane) * 16);
      k0 = wmma_bf16(a, b_frag(wr_pk + w0), k0);
      k1 = wmma_bf16(a, b_frag(wr_pk + w1), k1);
      v0 = wmma_bf16(a, b_frag(wv_pk + w0), v0);
      v1 = wmma_bf16(a, b_frag(wv_pk + w1), v1);
    }

    // biases + per-lane partial Q.K over this wave's 32 channels
    float part[8];
#pragma unroll
    for (int i = 0; i < 8; ++i) {
      k0[i] += kb0; k1[i] += kb1;
      v0[i] += vb0; v1[i] += vb1;
      part[i] = pq0[i] * k0[i] + pq1[i] * k1[i];
    }
    // reduce across the 16 lanes (N dimension) of each half-group
#pragma unroll
    for (int i = 0; i < 8; ++i) {
#pragma unroll
      for (int mk = 1; mk < 16; mk <<= 1)
        part[i] += __shfl_xor(part[i], mk, 32);
    }
    // cross-wave accumulation of the full 256-channel dot via LDS float atomics
    if (laneM == 0) {
#pragma unroll
      for (int i = 0; i < 8; ++i)
        atomicAdd(&lds_logit[r][i + 8 * half], part[i]);
    }
    __syncthreads();  // full logits for this r ready

    // online softmax update (rows i + 8*half)
    const float scale = 0.0625f;  // 1/sqrt(OUT)
#pragma unroll
    for (int i = 0; i < 8; ++i) {
      const float l  = lds_logit[r][i + 8 * half] * scale;
      const float mn = fmaxf(m[i], l);
      const float c  = __expf(m[i] - mn);
      const float w  = __expf(l - mn);
      s[i] = s[i] * c + w;
      m[i] = mn;
      o0[i] = o0[i] * c + w * v0[i];
      o1[i] = o1[i] * c + w * v1[i];
    }
  }

  // ---------------- out = pep_query + softmax-weighted value ----------------
#pragma unroll
  for (int i = 0; i < 8; ++i) {
    const int row = i + 8 * half;
    const size_t base = ((size_t)b * N_PEP_ + n0 + row) * OUT_;
    const float inv = 1.0f / s[i];
    out[base + c0 + laneM] = pq0[i] + o0[i] * inv;
    out[base + c1 + laneM] = pq1[i] + o1[i] * inv;
  }
}

extern "C" void kernel_launch(void* const* d_in, const int* in_sizes, int n_in,
                              void* d_out, int out_size, void* d_ws, size_t ws_size,
                              hipStream_t stream) {
  const float* rec  = (const float*)d_in[0];
  const float* pep  = (const float*)d_in[1];
  const float* edge = (const float*)d_in[2];
  const float* Wp_w = (const float*)d_in[3];
  const float* Wp_b = (const float*)d_in[4];
  const float* Wr_w = (const float*)d_in[5];
  const float* Wr_b = (const float*)d_in[6];
  const float* Wv_w = (const float*)d_in[7];
  const float* Wv_b = (const float*)d_in[8];
  float* out  = (float*)d_out;
  __bf16* wpk = (__bf16*)d_ws;   // 640 KB of packed bf16 weights

  hipLaunchKernelGGL(pack_weights_kernel, dim3(80), dim3(256), 0, stream,
                     Wp_w, Wr_w, Wv_w, wpk);

  dim3 grid(B_DIM * (N_PEP_ / 16));  // 1024 workgroups, one per 16 pep rows
  dim3 block(256);                   // 8 wave32
  hipLaunchKernelGGL(ga_wmma_kernel, grid, block, 0, stream,
                     rec, pep, edge, wpk, Wp_b, Wr_b, Wv_b, out);
}
